// MultiHeadAttention_55722905698860
// MI455X (gfx1250) — matrix operand order repaired and checked by reading, not compiled, not run on hardware
//
#include <hip/hip_runtime.h>
#include <hip/hip_bf16.h>

// ---------------------------------------------------------------------------
// MHA block for MI455X (gfx1250, wave32, WMMA bf16 16x16x32, f32 accum).
//   B=2, S=2048, D=1024, H=16, dk=64.
// Pipeline: 3x proj GEMM (f32->bf16 head layout) -> flash attention
//           -> O-proj GEMM (bf16 A -> f32) -> residual + layernorm.
// CDNA5 paths: v_wmma_f32_16x16x32_bf16, TDM tensor_load_to_lds (+ pad) for
// pure-copy LDS staging, s_wait_tensorcnt, ds_load_b128 fragment loads.
// Workspace (d_ws): Qh/Kh/Vh bf16 [B][H][S][64] (3x8MB), Ctx bf16 [B][S][D]
// (8MB), O f32 [B][S][D] (16MB) = 48 MB total.
// ---------------------------------------------------------------------------

#define MHA_B   2
#define MHA_S   2048
#define MHA_D   1024
#define MHA_H   16
#define MHA_DK  64
#define MHA_M   (MHA_B * MHA_S)   // 4096 rows

typedef __attribute__((ext_vector_type(16))) __bf16        v16bf;
typedef __attribute__((ext_vector_type(8)))  float         v8f;
typedef __attribute__((ext_vector_type(4)))  unsigned int  u32x4;
typedef __attribute__((ext_vector_type(8)))  unsigned int  u32x8;

union FragU { int4 q[2]; v16bf v; };
union U8    { int4 q; unsigned short h[8]; };

__device__ __forceinline__ unsigned short f2bf(float x) {
    unsigned u = __float_as_uint(x);
    unsigned r = u + 0x7FFFu + ((u >> 16) & 1u);   // round-to-nearest-even
    return (unsigned short)(r >> 16);
}

__device__ __forceinline__ v8f wmma_bf16(v16bf a, v16bf b, v8f c) {
    return __builtin_amdgcn_wmma_f32_16x16x32_bf16(false, a, false, b,
                                                   (short)0, c, false, false);
}

// ---------------------------------------------------------------------------
// TDM: async 2-D bf16 tile copy global -> LDS with row padding.
//   tile_w elements per row (2B each), tile_h rows, g_stride = global row
//   stride in elements. pad_int/pad_amt per D# group1 encodings:
//     pad interval (1<<code) 8-byte units, pad amount (code+1) DWORDs.
// 2-D tensor => descriptor groups 0 and 1 only (VADDR2/3 NULL form).
// Issue from ONE wave; EXEC is ignored by TDM. Tracked by TENSORcnt.
// ---------------------------------------------------------------------------
__device__ __forceinline__ void tdm_load_2d_bf16(unsigned lds_off, const void* gptr,
                                                 unsigned tile_w, unsigned tile_h,
                                                 unsigned g_stride,
                                                 unsigned pad_int, unsigned pad_amt) {
    unsigned long long ga = (unsigned long long)(size_t)gptr;
    u32x4 g0;
    g0[0] = 1u;                                           // count=1, user mode
    g0[1] = lds_off;                                      // lds_addr (bytes)
    g0[2] = (unsigned)(ga & 0xFFFFFFFFu);                 // global_addr[31:0]
    g0[3] = (unsigned)((ga >> 32) & 0x01FFFFFFu)          // global_addr[56:32]
          | (2u << 30);                                   // type = 2 ("image")
    u32x8 g1;
    g1[0] = (1u << 16)                                    // data_size = 2 bytes
          | (1u << 20)                                    // pad_enable
          | ((pad_int & 7u) << 22)                        // pad_interval
          | ((pad_amt & 127u) << 25);                     // pad_amount
    g1[1] = (tile_w & 0xFFFFu) << 16;                     // tensor_dim0[15:0]
    g1[2] = (tile_h & 0xFFFFu) << 16;                     // tensor_dim1[15:0]
    g1[3] = (tile_w & 0xFFFFu) << 16;                     // tile_dim0
    g1[4] = (tile_h & 0xFFFFu);                           // tile_dim1 (tile_dim2=0)
    g1[5] = g_stride;                                     // tensor_dim0_stride lo
    g1[6] = 0u;
    g1[7] = 0u;
    asm volatile("tensor_load_to_lds %0, %1" :: "s"(g0), "s"(g1) : "memory");
}

// A fragment (16x32 bf16), LDS row-major [m][k], stride in elements.
// ISA: lanes 0-15: M=lane, K = {0..7, 16..23}; lanes 16-31: K = {8..15, 24..31}.
__device__ __forceinline__ v16bf load_frag_a(const unsigned short* base, int stride,
                                             int m_base, int k_base) {
    const int lane = (int)(threadIdx.x & 31u);
    const int row  = m_base + (lane & 15);
    const int kh   = (lane < 16) ? 0 : 8;
    const unsigned short* p = base + row * stride + k_base + kh;
    FragU f;
    f.q[0] = *reinterpret_cast<const int4*>(p);        // K +0..7
    f.q[1] = *reinterpret_cast<const int4*>(p + 16);   // K +16..23
    return f.v;
}

// B fragment (32x16 bf16), LDS stored transposed [n][k], stride in elements.
// Layout: lanes 0-15: N=lane, K = {0..7, 16..23}; lanes 16-31: K = {8..15, 24..31}.
__device__ __forceinline__ v16bf load_frag_b(const unsigned short* base, int stride,
                                             int n_base, int k_base) {
    const int lane = (int)(threadIdx.x & 31u);
    const int col  = n_base + (lane & 15);
    const int kh   = (lane < 16) ? 0 : 8;
    const unsigned short* p = base + col * stride + k_base + kh;
    FragU f;
    f.q[0] = *reinterpret_cast<const int4*>(p);        // K +0..7
    f.q[1] = *reinterpret_cast<const int4*>(p + 16);   // K +16..23
    return f.v;
}

__device__ __forceinline__ float half_reduce_max(float v) {
#pragma unroll
    for (int off = 1; off < 16; off <<= 1)
        v = fmaxf(v, __shfl_xor(v, off, 32));
    return v;
}
__device__ __forceinline__ float half_reduce_sum(float v) {
#pragma unroll
    for (int off = 1; off < 16; off <<= 1)
        v += __shfl_xor(v, off, 32);
    return v;
}

// ---------------------------------------------------------------------------
// GEMM:  Out = A[4096x1024] * W[1024x1024] + bias
//   A_F32:    A is f32 (VALU convert staging); else bf16 (TDM staging)
//   HEAD_OUT: write bf16 to [B][H][S][64], else f32 row-major
// Block tile 128x128, BK=32; 8 waves, each 64x32 (4x2 WMMA tiles).
// ---------------------------------------------------------------------------
template <bool A_F32, bool HEAD_OUT>
__global__ __launch_bounds__(256) void gemm_bias_kernel(const void* __restrict__ A_,
                                                        const float* __restrict__ W,
                                                        const float* __restrict__ bias,
                                                        void* __restrict__ Out) {
    constexpr int K = MHA_D, N = MHA_D;
    __shared__ __align__(16) unsigned short As[128][40];  // 80B rows = 64 + 16 pad
    __shared__ __align__(16) unsigned short Bs[128][40];  // W transposed: [n][k]

    const int tid  = (int)threadIdx.x;
    const int lane = tid & 31;
    const int wave = tid >> 5;
    const int wrow = wave >> 2;   // 0..1
    const int wcol = wave & 3;    // 0..3
    const int m0 = (int)blockIdx.y * 128;
    const int n0 = (int)blockIdx.x * 128;
    const unsigned As_off = (unsigned)(size_t)&As[0][0];

    const v8f zero8 = {0.f, 0.f, 0.f, 0.f, 0.f, 0.f, 0.f, 0.f};
    v8f acc[4][2];
#pragma unroll
    for (int i = 0; i < 4; ++i)
#pragma unroll
        for (int j = 0; j < 2; ++j) acc[i][j] = zero8;

    for (int kt = 0; kt < K; kt += 32) {
        // ---- stage A tile (128 x 32) as bf16 ----
        if (A_F32) {
            const float* A = (const float*)A_;
#pragma unroll
            for (int it = 0; it < 4; ++it) {
                int slot = tid + it * 256;            // 1024 = 128 rows * 8 grp
                int row = slot >> 3, c = (slot & 7) * 4;
                float4 f = *reinterpret_cast<const float4*>(
                    A + (size_t)(m0 + row) * K + kt + c);
                As[row][c + 0] = f2bf(f.x);
                As[row][c + 1] = f2bf(f.y);
                As[row][c + 2] = f2bf(f.z);
                As[row][c + 3] = f2bf(f.w);
            }
        } else {
            // bf16 A: pure copy -> Tensor Data Mover. Rows of 64B (16 DWORDs
            // -> interval code 3), pad 16B (4 DWORDs -> amount code 3).
            if (wave == 0) {
                const unsigned short* A = (const unsigned short*)A_;
                tdm_load_2d_bf16(As_off, A + (size_t)m0 * K + kt,
                                 /*w*/32, /*h*/128, /*stride*/K, 3, 3);
            }
        }
        // ---- stage W tile (32 x 128) transposed into Bs[n][k] ----
#pragma unroll
        for (int it = 0; it < 4; ++it) {
            int slot = tid + it * 256;                // 1024 = 32 k * 32 grp
            int kk = slot >> 5, n = (slot & 31) * 4;
            float4 f = *reinterpret_cast<const float4*>(
                W + (size_t)(kt + kk) * N + n0 + n);
            Bs[n + 0][kk] = f2bf(f.x);
            Bs[n + 1][kk] = f2bf(f.y);
            Bs[n + 2][kk] = f2bf(f.z);
            Bs[n + 3][kk] = f2bf(f.w);
        }
        if (!A_F32 && wave == 0)
            __builtin_amdgcn_s_wait_tensorcnt(0);
        __syncthreads();

        v16bf afr[4], bfr[2];
#pragma unroll
        for (int i = 0; i < 4; ++i)
            afr[i] = load_frag_a(&As[0][0], 40, wrow * 64 + i * 16, 0);
#pragma unroll
        for (int j = 0; j < 2; ++j)
            bfr[j] = load_frag_b(&Bs[0][0], 40, wcol * 32 + j * 16, 0);
#pragma unroll
        for (int i = 0; i < 4; ++i)
#pragma unroll
            for (int j = 0; j < 2; ++j)
                acc[i][j] = wmma_bf16(afr[i], bfr[j], acc[i][j]);
        __syncthreads();
    }

    // ---- epilogue: bias add + store ----
#pragma unroll
    for (int i = 0; i < 4; ++i) {
#pragma unroll
        for (int j = 0; j < 2; ++j) {
            int mbase = m0 + wrow * 64 + i * 16 + ((lane >> 4) << 3);
            int ncol  = n0 + wcol * 32 + j * 16 + (lane & 15);
            float b = bias[ncol];
#pragma unroll
            for (int r = 0; r < 8; ++r) {
                int m = mbase + r;
                float v = acc[i][j][r] + b;
                if (HEAD_OUT) {
                    int bb = m >> 11, s = m & 2047;          // S = 2048
                    int h  = ncol >> 6, d = ncol & 63;       // dk = 64
                    size_t idx = (((size_t)(bb * MHA_H + h) * MHA_S + s) << 6) + d;
                    ((unsigned short*)Out)[idx] = f2bf(v);
                } else {
                    ((float*)Out)[(size_t)m * N + ncol] = v;
                }
            }
        }
    }
}

// ---------------------------------------------------------------------------
// Flash attention: grid (S/128, B*H). Block = 256 thr = 8 waves.
// Wave owns 16 query rows x dk=64; KV tiles of 64; online softmax (exp2
// domain, 1/sqrt(dk)*log2e folded into the exponentials, so Q/K staging are
// pure copies handled by the TDM).
// ---------------------------------------------------------------------------
__global__ __launch_bounds__(256) void flash_attn_kernel(const unsigned short* __restrict__ Qh,
                                                         const unsigned short* __restrict__ Kh,
                                                         const unsigned short* __restrict__ Vh,
                                                         unsigned short* __restrict__ Ctx) {
    __shared__ __align__(16) unsigned short Qs[128][72];   // [q_row][d]
    __shared__ __align__(16) unsigned short Ks[64][72];    // [kv_row][d]
    __shared__ __align__(16) unsigned short Vt[64][72];    // [d][kv_row]
    __shared__ __align__(16) unsigned short Ps[8][16][72]; // per-wave P tile

    const int tid  = (int)threadIdx.x;
    const int lane = tid & 31;
    const int wave = tid >> 5;
    const int qtile = (int)blockIdx.x;           // 0..15
    const int bh    = (int)blockIdx.y;           // 0..31
    const int b = bh >> 4, h = bh & 15;

    const float CSCALE = 0.125f * 1.44269504088896f;   // 1/sqrt(dk) * log2(e)
    const unsigned Qs_off = (unsigned)(size_t)&Qs[0][0];
    const unsigned Ks_off = (unsigned)(size_t)&Ks[0][0];

    // ---- Q tile (128 x 64 bf16, contiguous) via TDM; rows 128B -> interval
    // code 4, pad 16B -> amount code 3 (LDS stride 72 elems = 144B) ----
    const unsigned short* Qbase = Qh + ((size_t)bh * MHA_S + (size_t)qtile * 128) * MHA_DK;
    if (wave == 0)
        tdm_load_2d_bf16(Qs_off, Qbase, /*w*/64, /*h*/128, /*stride*/64, 4, 3);

    const v8f zero8 = {0.f, 0.f, 0.f, 0.f, 0.f, 0.f, 0.f, 0.f};
    v8f o[4];
#pragma unroll
    for (int j = 0; j < 4; ++j) o[j] = zero8;
    float m_st[8], l_st[8];
#pragma unroll
    for (int i = 0; i < 8; ++i) { m_st[i] = -1e30f; l_st[i] = 0.f; }

    const unsigned short* Kbh = Kh + (size_t)bh * MHA_S * MHA_DK;
    const unsigned short* Vbh = Vh + (size_t)bh * MHA_S * MHA_DK;

    for (int t = 0; t < MHA_S / 64; ++t) {
        __syncthreads();   // previous iteration done reading Ks/Vt
        const unsigned short* Kt = Kbh + (size_t)t * 64 * MHA_DK;
        const unsigned short* Vb = Vbh + (size_t)t * 64 * MHA_DK;
        // K tile: pure copy -> TDM (64x64 bf16)
        if (wave == 0)
            tdm_load_2d_bf16(Ks_off, Kt, /*w*/64, /*h*/64, /*stride*/64, 4, 3);
        // V tile: transpose staging on the VALU path
#pragma unroll
        for (int it = 0; it < 2; ++it) {
            int slot = tid + it * 256;           // 512 = 64 rows * 8 grp
            int row = slot >> 3, c = (slot & 7) * 8;
            U8 u; u.q = *reinterpret_cast<const int4*>(Vb + row * MHA_DK + c);
#pragma unroll
            for (int e = 0; e < 8; ++e) Vt[c + e][row] = u.h[e];
        }
        if (wave == 0)
            __builtin_amdgcn_s_wait_tensorcnt(0);  // Q (t=0) + K complete
        __syncthreads();

        // ---- S = Q @ K^T (raw scores) : 16 x 64 per wave ----
        v8f s[4];
#pragma unroll
        for (int j = 0; j < 4; ++j) s[j] = zero8;
#pragma unroll
        for (int ks = 0; ks < 2; ++ks) {
            v16bf a = load_frag_a(&Qs[0][0], 72, wave * 16, ks * 32);
#pragma unroll
            for (int j = 0; j < 4; ++j) {
                v16bf bf = load_frag_b(&Ks[0][0], 72, j * 16, ks * 32);
                s[j] = wmma_bf16(a, bf, s[j]);
            }
        }

        // ---- online softmax; scale folded into exp2 arguments ----
#pragma unroll
        for (int i = 0; i < 8; ++i) {
            float v = fmaxf(fmaxf(s[0][i], s[1][i]), fmaxf(s[2][i], s[3][i]));
            v = half_reduce_max(v);
            float mn = fmaxf(m_st[i], v);
            float sc = exp2f((m_st[i] - mn) * CSCALE);
            m_st[i] = mn;
            l_st[i] *= sc;
#pragma unroll
            for (int j = 0; j < 4; ++j) o[j][i] *= sc;
            float rs = 0.f;
#pragma unroll
            for (int j = 0; j < 4; ++j) {
                float p = exp2f((s[j][i] - mn) * CSCALE);
                s[j][i] = p;
                rs += p;
            }
            l_st[i] += half_reduce_sum(rs);
        }

        // ---- P -> LDS (bf16) to re-fragment into A layout ----
#pragma unroll
        for (int j = 0; j < 4; ++j)
#pragma unroll
            for (int i = 0; i < 8; ++i) {
                int row = i + ((lane >> 4) << 3);
                int col = j * 16 + (lane & 15);
                Ps[wave][row][col] = f2bf(s[j][i]);
            }
        __syncthreads();

        // ---- O += P @ V ----
#pragma unroll
        for (int ks = 0; ks < 2; ++ks) {
            v16bf a = load_frag_a(&Ps[wave][0][0], 72, 0, ks * 32);
#pragma unroll
            for (int j = 0; j < 4; ++j) {
                v16bf bf = load_frag_b(&Vt[0][0], 72, j * 16, ks * 32);
                o[j] = wmma_bf16(a, bf, o[j]);
            }
        }
    }

    // ---- normalize and scatter to Ctx [B][S][D] (bf16) ----
#pragma unroll
    for (int i = 0; i < 8; ++i) {
        float inv = 1.0f / l_st[i];
        int srow = qtile * 128 + wave * 16 + i + ((lane >> 4) << 3);
#pragma unroll
        for (int j = 0; j < 4; ++j) {
            int d = j * 16 + (lane & 15);
            size_t idx = ((size_t)(b * MHA_S + srow)) * MHA_D + h * MHA_DK + d;
            Ctx[idx] = f2bf(o[j][i] * inv);
        }
    }
}

// ---------------------------------------------------------------------------
// out = layernorm(O + residual) * gamma + beta      (one block per row)
// ---------------------------------------------------------------------------
__global__ __launch_bounds__(256) void add_layernorm_kernel(const float* __restrict__ O,
                                                            const float* __restrict__ X,
                                                            const float* __restrict__ gamma,
                                                            const float* __restrict__ beta,
                                                            float* __restrict__ out) {
    __shared__ float red[256];
    const int row = (int)blockIdx.x;
    const float* o = O + (size_t)row * MHA_D;
    const float* x = X + (size_t)row * MHA_D;

    float v[4], s = 0.f;
#pragma unroll
    for (int i = 0; i < 4; ++i) {
        int c = (int)threadIdx.x + i * 256;
        v[i] = o[c] + x[c];
        s += v[i];
    }
    red[threadIdx.x] = s;
    __syncthreads();
    for (int off = 128; off > 0; off >>= 1) {
        if ((int)threadIdx.x < off) red[threadIdx.x] += red[threadIdx.x + off];
        __syncthreads();
    }
    float mean = red[0] * (1.0f / MHA_D);
    __syncthreads();

    float s2 = 0.f;
#pragma unroll
    for (int i = 0; i < 4; ++i) {
        float d = v[i] - mean;
        s2 += d * d;
    }
    red[threadIdx.x] = s2;
    __syncthreads();
    for (int off = 128; off > 0; off >>= 1) {
        if ((int)threadIdx.x < off) red[threadIdx.x] += red[threadIdx.x + off];
        __syncthreads();
    }
    float rstd = rsqrtf(red[0] * (1.0f / MHA_D) + 1e-5f);

#pragma unroll
    for (int i = 0; i < 4; ++i) {
        int c = (int)threadIdx.x + i * 256;
        out[(size_t)row * MHA_D + c] = (v[i] - mean) * rstd * gamma[c] + beta[c];
    }
}

// ---------------------------------------------------------------------------
extern "C" void kernel_launch(void* const* d_in, const int* in_sizes, int n_in,
                              void* d_out, int out_size, void* d_ws, size_t ws_size,
                              hipStream_t stream) {
    (void)in_sizes; (void)n_in; (void)out_size; (void)ws_size;

    const float* query = (const float*)d_in[0];
    const float* key   = (const float*)d_in[1];
    const float* value = (const float*)d_in[2];
    const float* Wq = (const float*)d_in[3];
    const float* bq = (const float*)d_in[4];
    const float* Wk = (const float*)d_in[5];
    const float* bk = (const float*)d_in[6];
    const float* Wv = (const float*)d_in[7];
    const float* bv = (const float*)d_in[8];
    const float* Wo = (const float*)d_in[9];
    const float* bo = (const float*)d_in[10];
    const float* ln_gamma = (const float*)d_in[11];
    const float* ln_beta  = (const float*)d_in[12];

    const size_t szQ = (size_t)MHA_M * MHA_D;         // elements
    unsigned short* Qh  = (unsigned short*)d_ws;
    unsigned short* Kh  = Qh + szQ;
    unsigned short* Vh  = Kh + szQ;
    unsigned short* Ctx = Vh + szQ;
    float*          Ob  = (float*)(Ctx + szQ);

    dim3 gemm_grid(MHA_D / 128, MHA_M / 128);         // (8, 32)
    dim3 block(256);

    gemm_bias_kernel<true, true><<<gemm_grid, block, 0, stream>>>(query, Wq, bq, Qh);
    gemm_bias_kernel<true, true><<<gemm_grid, block, 0, stream>>>(key,   Wk, bk, Kh);
    gemm_bias_kernel<true, true><<<gemm_grid, block, 0, stream>>>(value, Wv, bv, Vh);

    dim3 fa_grid(MHA_S / 128, MHA_B * MHA_H);         // (16, 32)
    flash_attn_kernel<<<fa_grid, block, 0, stream>>>(Qh, Kh, Vh, Ctx);

    gemm_bias_kernel<false, false><<<gemm_grid, block, 0, stream>>>(Ctx, Wo, bo, Ob);

    add_layernorm_kernel<<<MHA_M, block, 0, stream>>>(Ob, query, ln_gamma, ln_beta,
                                                      (float*)d_out);
}
